// SnowCLRLoss_61366492725250
// MI455X (gfx1250) — matrix operand order repaired and checked by reading, not compiled, not run on hardware
//
#include <hip/hip_runtime.h>
#include <hip/hip_bf16.h>

typedef __attribute__((ext_vector_type(16))) __bf16 bf16x16;
typedef __attribute__((ext_vector_type(8)))  float  f32x8;
typedef __attribute__((ext_vector_type(4)))  int    i32x4;

#define FDIM 512
#define BDIM 256
#define QSIZE 1024
#define BSTRIDE 80              // 64B of bf16 data + 16B pad -> conflict-min ds_load_b128
#define SLAB (128*BSTRIDE)      // one k-slice: 128 cols x 80B = 10KB

#if defined(__AMDGCN__) && __has_builtin(__builtin_amdgcn_global_load_async_to_lds_b128) && __has_builtin(__builtin_amdgcn_s_wait_asynccnt)
#define USE_ASYNC_G2L 1
#else
#define USE_ASYNC_G2L 0
#endif

#if USE_ASYNC_G2L
typedef __attribute__((address_space(1))) i32x4 gmem_i32x4;
typedef __attribute__((address_space(3))) i32x4 lds_i32x4;
__device__ __forceinline__ void g2l_b128(const void* g, void* l) {
    // GLOBAL_LOAD_ASYNC_TO_LDS_B128: LDS[l] = MEM[g], tracked by ASYNCcnt
    __builtin_amdgcn_global_load_async_to_lds_b128(
        (gmem_i32x4*)g, (lds_i32x4*)l, 0, 0);
}
#endif

// ---------------------------------------------------------------- utilities
__device__ __forceinline__ unsigned fkey(float f) {
    unsigned u = __float_as_uint(f);
    return (u & 0x80000000u) ? ~u : (u | 0x80000000u);
}

// block-wide sum over 256 threads (8 waves of 32); result broadcast to all.
__device__ __forceinline__ float blockReduceSum(float v, float* sbuf) {
    const int lane = threadIdx.x & 31;
    const int wave = threadIdx.x >> 5;
    #pragma unroll
    for (int o = 16; o > 0; o >>= 1) v += __shfl_xor(v, o);
    if (lane == 0) sbuf[wave] = v;
    __syncthreads();
    float r = (threadIdx.x < 8) ? sbuf[threadIdx.x] : 0.0f;
    if (wave == 0) {
        #pragma unroll
        for (int o = 4; o > 0; o >>= 1) r += __shfl_xor(r, o);
        if (lane == 0) sbuf[0] = r;
    }
    __syncthreads();
    float out = sbuf[0];
    __syncthreads();
    return out;
}

// ------------------------------------------------ kernel 1: prep org / aug
__global__ __launch_bounds__(256)
void prep_kernel(const float* __restrict__ V, const float* __restrict__ L,
                 float* __restrict__ aug, unsigned short* __restrict__ orgb_u,
                 unsigned long long* __restrict__ gmax, float* __restrict__ out) {
    const int b = blockIdx.x, tid = threadIdx.x;
    __shared__ float sred[8];
    __bf16* orgb = (__bf16*)orgb_u;
    const size_t BF = (size_t)BDIM * FDIM;

    float l0 = L[(size_t)b*FDIM + tid];
    float l1 = L[(size_t)b*FDIM + tid + 256];
    float sL = blockReduceSum(l0*l0 + l1*l1, sred);
    float ln = fmaxf(sqrtf(sL), 1e-12f);

    #pragma unroll
    for (int a = 0; a < 3; ++a) {
        float v0 = V[a*BF + (size_t)b*FDIM + tid];
        float v1 = V[a*BF + (size_t)b*FDIM + tid + 256];
        float q0 = v0*l0, q1 = v1*l1;
        float sq = blockReduceSum(q0*q0 + q1*q1, sred);
        float denom = fmaxf(sqrtf(sq), 1e-12f*ln);
        float o0 = q0/denom, o1 = q1/denom;
        if (a == 0) {
            orgb[(size_t)b*FDIM + tid]       = (__bf16)o0;
            orgb[(size_t)b*FDIM + tid + 256] = (__bf16)o1;
        } else {
            aug[(size_t)(a-1)*BF + (size_t)b*FDIM + tid]       = o0;
            aug[(size_t)(a-1)*BF + (size_t)b*FDIM + tid + 256] = o1;
        }
    }
    if (tid == 0) {
        gmax[b] = 0ULL;
        if (b == 0) out[0] = 0.0f;
    }
}

// --------------------------------- kernel 2: bf16 WMMA GEMM + fused argmax
// Block: 8 waves x 16 support rows = 128 M-rows, all 256 batch cols (2 passes
// of 128).  Per k-step the block stages the shared B slice (128 cols x 32 k,
// bf16) into LDS via async global->LDS, then every wave loads all 8 B
// fragments up front (clause-able ds_load_b128) and issues 8 back-to-back
// v_wmma_f32_16x16x32_bf16.
__global__ __launch_bounds__(256)
void sim_argmax_kernel(const float* __restrict__ support,
                       const unsigned short* __restrict__ orgb_u,
                       unsigned long long* __restrict__ gmax, int S) {
    __shared__ unsigned long long smax[256];
    __shared__ __align__(16) char bslice[2*SLAB];   // 20KB double buffer
    const int tid = threadIdx.x;
    smax[tid] = 0ULL;

    const __bf16* orgb = (const __bf16*)orgb_u;
    const int lane = tid & 31;
    const int wave = tid >> 5;
    const int half = lane >> 4;       // A: K 0..15 vs 16..31; B: K 0..15 vs 16..31
    const int lc   = lane & 15;
    const int mtile = blockIdx.x * 128 + wave * 16;
    const bool active = (mtile < S);  // wave-uniform: EXEC all-ones inside

    int mrow = mtile + lc;
    if (mrow >= S) mrow = S - 1;      // clamp (junk rows excluded at reduction)
    const float* arow = support + (size_t)mrow * FDIM;

    // cooperative slice-copy coordinates: thread -> (column c, 32B half h)
    const int c = tid >> 1;
    const int h = tid & 1;
    char* ldst = bslice + c*BSTRIDE + h*32;

    __syncthreads();                  // smax init visible

    #pragma unroll 1
    for (int ng = 0; ng < 2; ++ng) {
        const char* gsrc = (const char*)(orgb + (size_t)(ng*128 + c)*FDIM + h*16);
        f32x8 zero = {};
        f32x8 acc[8];
        #pragma unroll
        for (int j = 0; j < 8; ++j) acc[j] = zero;

        auto compute = [&](int k0, int buf) {
            __builtin_prefetch(arow + k0 + 64, 0, 3);
            // A fragment: global f32 -> bf16
            const float4* pa = (const float4*)(arow + k0 + half*16);
            float4 f0 = pa[0], f1 = pa[1];
            const float4* pb = (const float4*)(arow + k0 + half*16 + 8);
            float4 g0 = pb[0], g1 = pb[1];
            // all 8 B fragments up front: 16 clause-able ds_load_b128,
            // waits pipeline instead of one s_wait_dscnt 0 per WMMA
            const char* base = bslice + buf*SLAB;
            union Frag { uint4 u[2]; bf16x16 v; };
            Frag bb[8];
            #pragma unroll
            for (int j = 0; j < 8; ++j) {
                const char* bp = base + (j*16 + lc)*BSTRIDE + half*32;
                bb[j].u[0] = *(const uint4*)bp;
                bb[j].u[1] = *(const uint4*)(bp + 16);
            }
            bf16x16 a;
            a[0]=(__bf16)f0.x;  a[1]=(__bf16)f0.y;  a[2]=(__bf16)f0.z;  a[3]=(__bf16)f0.w;
            a[4]=(__bf16)f1.x;  a[5]=(__bf16)f1.y;  a[6]=(__bf16)f1.z;  a[7]=(__bf16)f1.w;
            a[8]=(__bf16)g0.x;  a[9]=(__bf16)g0.y;  a[10]=(__bf16)g0.z; a[11]=(__bf16)g0.w;
            a[12]=(__bf16)g1.x; a[13]=(__bf16)g1.y; a[14]=(__bf16)g1.z; a[15]=(__bf16)g1.w;
            #pragma unroll
            for (int j = 0; j < 8; ++j) {
                acc[j] = __builtin_amdgcn_wmma_f32_16x16x32_bf16(
                    false, a, false, bb[j].v, (short)0, acc[j], false, false);
            }
        };

#if USE_ASYNC_G2L
        // slice 0
        g2l_b128(gsrc,      ldst);
        g2l_b128(gsrc + 16, ldst + 16);
        #pragma unroll 1
        for (int ks = 0; ks < 16; ++ks) {
            const int k0 = ks * 32;
            if (ks < 15) {              // issue slice ks+1 into the other buffer
                const char* gn = gsrc + (size_t)(k0 + 32) * 2;
                char* ln = ldst + ((ks + 1) & 1) * SLAB;
                g2l_b128(gn,      ln);
                g2l_b128(gn + 16, ln + 16);
                __builtin_amdgcn_s_wait_asynccnt(2);   // slice ks resident, ks+1 in flight
            } else {
                __builtin_amdgcn_s_wait_asynccnt(0);
            }
            __syncthreads();            // every wave's portion of slice ks visible
            if (active) compute(k0, ks & 1);
            __syncthreads();            // done reading before slice ks+2 overwrites
        }
#else
        // register-staged double buffer fallback
        uint4 r0, r1;
        {
            const uint4* gs = (const uint4*)gsrc;
            r0 = gs[0]; r1 = gs[1];
            *(uint4*)ldst = r0; *(uint4*)(ldst + 16) = r1;
        }
        __syncthreads();
        #pragma unroll 1
        for (int ks = 0; ks < 16; ++ks) {
            const int k0 = ks * 32;
            if (ks < 15) {
                const uint4* gs = (const uint4*)(gsrc + (size_t)(k0 + 32) * 2);
                r0 = gs[0]; r1 = gs[1];
            }
            if (active) compute(k0, ks & 1);
            if (ks < 15) {
                char* ln = ldst + ((ks + 1) & 1) * SLAB;
                *(uint4*)ln = r0; *(uint4*)(ln + 16) = r1;
            }
            __syncthreads();
        }
#endif

        // fused per-column argmax.  C layout: lane holds col n = tile*16+lc,
        // rows M = mtile + half*8 + r in VGPR r.
        if (active) {
            #pragma unroll
            for (int j = 0; j < 8; ++j) {
                float best = -3.0e38f; int bi = mtile;
                #pragma unroll
                for (int r = 0; r < 8; ++r) {
                    int m = mtile + half*8 + r;
                    float v = acc[j][r];
                    bool ok = (m < S) && (v > best);
                    best = ok ? v : best;
                    bi   = ok ? m : bi;
                }
                unsigned long long pk =
                    ((unsigned long long)fkey(best) << 32) |
                    (unsigned long long)(0xFFFFFFFFu - (unsigned)bi);
                unsigned hi = __shfl_xor((unsigned)(pk >> 32), 16);
                unsigned lo = __shfl_xor((unsigned)(pk & 0xFFFFFFFFu), 16);
                unsigned long long other = ((unsigned long long)hi << 32) | lo;
                if (other > pk) pk = other;
                if (lane < 16)
                    atomicMax(&smax[ng*128 + j*16 + lc], pk);
            }
        }
    }
    __syncthreads();
    unsigned long long v = smax[tid];
    if (v) atomicMax(&gmax[tid], v);
}

// ---------------------------------------- kernel 3: gather + normalize nn
__global__ __launch_bounds__(256)
void nn_kernel(const float* __restrict__ support,
               const unsigned long long* __restrict__ gmax,
               float* __restrict__ nn) {
    const int b = blockIdx.x, tid = threadIdx.x;
    __shared__ float sred[8];
    unsigned long long key = gmax[b];
    int idx = (int)(0xFFFFFFFFu - (unsigned)(key & 0xFFFFFFFFu));
    const float* row = support + (size_t)idx * FDIM;
    float x0 = row[tid], x1 = row[tid + 256];
    float sq = blockReduceSum(x0*x0 + x1*x1, sred);
    float d = fmaxf(sqrtf(sq), 1e-12f);
    nn[(size_t)b*FDIM + tid]       = x0 / d;
    nn[(size_t)b*FDIM + tid + 256] = x1 / d;
}

// -------------------- kernel 4: queue selection + denominators + loss term
__global__ __launch_bounds__(256)
void loss_kernel(const float* __restrict__ support,
                 const float* __restrict__ sgps,
                 const float* __restrict__ gps,
                 const float* __restrict__ aug,
                 const float* __restrict__ nn,
                 float* __restrict__ out, int S) {
    const int b = blockIdx.x, tid = threadIdx.x;
    const int lane = tid & 31, wave = tid >> 5;
    const size_t BF = (size_t)BDIM * FDIM;

    __shared__ float snn[FDIM];
    __shared__ int   sidx[QSIZE];
    __shared__ int   scount;
    __shared__ int   swcnt[8];
    __shared__ float sred[8];

    snn[tid]       = nn[(size_t)b*FDIM + tid];
    snn[tid + 256] = nn[(size_t)b*FDIM + tid + 256];
    if (tid == 0) scount = 0;
    __syncthreads();

    float p0 = aug[(size_t)b*FDIM + tid]            * snn[tid]
             + aug[(size_t)b*FDIM + tid + 256]      * snn[tid + 256];
    float p1 = aug[BF + (size_t)b*FDIM + tid]       * snn[tid]
             + aug[BF + (size_t)b*FDIM + tid + 256] * snn[tid + 256];
    float d0 = blockReduceSum(p0, sred);
    float d1 = blockReduceSum(p1, sred);
    float batch_den = expf(d0 * 10.0f) + expf(d1 * 10.0f);

    const float D2R = 0.017453292519943295f;
    float blat = gps[2*b] * D2R, blon = gps[2*b + 1] * D2R;
    float cosb = cosf(blat);

    for (int s0 = 0; s0 < S; s0 += 256) {
        int s = s0 + tid;
        bool far = false;
        if (s < S) {
            float lat = sgps[2*s] * D2R, lon = sgps[2*s + 1] * D2R;
            float sa = sinf(0.5f * (lat - blat));
            float sb = sinf(0.5f * (lon - blon));
            float hh = sa*sa + cosb * cosf(lat) * sb*sb;
            hh = fminf(fmaxf(hh, 0.0f), 1.0f);
            float dist = 2.0f * 6371.0088f * asinf(sqrtf(hh));
            far = dist > 25.0f;
        }
        unsigned mm = (unsigned)__ballot(far);
        if (lane == 0) swcnt[wave] = __popc(mm);
        __syncthreads();
        int base = scount;
        int off = 0;
        #pragma unroll
        for (int w = 0; w < 8; ++w) off += (w < wave) ? swcnt[w] : 0;
        int pos = base + off + __popc(mm & ((1u << lane) - 1u));
        if (far && pos < QSIZE) sidx[pos] = s;
        __syncthreads();
        if (tid == 0) {
            int tot = 0;
            #pragma unroll
            for (int w = 0; w < 8; ++w) tot += swcnt[w];
            scount = base + tot;
        }
        __syncthreads();
        if (scount >= QSIZE) break;   // uniform
    }

    int cnt = scount; if (cnt > QSIZE) cnt = QSIZE;

    float acc = 0.0f;
    for (int q = wave; q < cnt; q += 8) {
        const float* row = support + (size_t)sidx[q] * FDIM;
        float dot = 0.0f, sq = 0.0f;
        #pragma unroll 4
        for (int i = 0; i < 16; ++i) {
            float x = row[lane + 32*i];
            dot = fmaf(x, snn[lane + 32*i], dot);
            sq  = fmaf(x, x, sq);
        }
        #pragma unroll
        for (int o = 16; o > 0; o >>= 1) { dot += __shfl_xor(dot, o); sq += __shfl_xor(sq, o); }
        if (lane == 0) acc += expf(dot / fmaxf(sqrtf(sq), 1e-12f) * 10.0f);
    }
    if (lane == 0) sred[wave] = acc;
    __syncthreads();
    if (tid == 0) {
        float qden = (float)(QSIZE - cnt);   // zero rows contribute exp(0)=1
        #pragma unroll
        for (int w = 0; w < 8; ++w) qden += sred[w];
        float term = -batch_den / (batch_den + qden);
        atomicAdd(out, term * (1.0f / 256.0f));
    }
}

// ------------------------------------------------------------------ launch
extern "C" void kernel_launch(void* const* d_in, const int* in_sizes, int n_in,
                              void* d_out, int out_size, void* d_ws, size_t ws_size,
                              hipStream_t stream) {
    const float* V       = (const float*)d_in[0];   // (3,256,512)
    const float* L       = (const float*)d_in[1];   // (256,512)
    const float* gps     = (const float*)d_in[2];   // (256,2)
    const float* support = (const float*)d_in[3];   // (S,512)
    const float* sgps    = (const float*)d_in[4];   // (S,2)
    const int S = in_sizes[3] / FDIM;

    char* ws = (char*)d_ws;
    float*              aug   = (float*)(ws);                        // 1,048,576 B
    float*              nn    = (float*)(ws + 1048576);              //   524,288 B
    unsigned short*     orgb  = (unsigned short*)(ws + 1572864);     //   262,144 B
    unsigned long long* gmax  = (unsigned long long*)(ws + 1835008); //     2,048 B
    float* out = (float*)d_out;

    prep_kernel<<<BDIM, 256, 0, stream>>>(V, L, aug, orgb, gmax, out);
    int nblk = (S + 127) / 128;
    sim_argmax_kernel<<<nblk, 256, 0, stream>>>(support, orgb, gmax, S);
    nn_kernel<<<BDIM, 256, 0, stream>>>(support, gmax, nn);
    loss_kernel<<<BDIM, 256, 0, stream>>>(support, sgps, gps, aug, nn, out, S);
}
